// FlashPhiAttention_54683523612686
// MI455X (gfx1250) — matrix operand order repaired and checked by reading, not compiled, not run on hardware
//
#include <hip/hip_runtime.h>
#include <hip/hip_bf16.h>

// ---------------- Types & helpers ----------------
typedef __attribute__((ext_vector_type(8)))  __bf16 bfx8;
typedef __attribute__((ext_vector_type(16))) __bf16 bfx16;
typedef __attribute__((ext_vector_type(8)))  float  fx8;
typedef int v4i __attribute__((vector_size(4 * sizeof(int))));

#define DEV static __device__ __forceinline__

constexpr int Bc  = 4;
constexpr int Sc  = 1024;
constexpr int Tc  = Bc * Sc;          // 4096
constexpr int Hc  = 2560;
constexpr int NHc = 32;
constexpr int HSc = 80;
constexpr int ROTc = 32;              // 0.4 * 80
constexpr int DP  = 96;               // head dim padded to multiple of 32
constexpr int QKV_N = 3 * NHc * HSc;  // 7680
constexpr float SCALEc = 0.11180339887498949f; // 80^-0.5

DEV unsigned short f2bf(float f) {
  unsigned int u = __float_as_uint(f);
  u += 0x7FFFu + ((u >> 16) & 1u);        // round to nearest even
  return (unsigned short)(u >> 16);
}
DEV float bf2f(unsigned short h) { return __uint_as_float(((unsigned int)h) << 16); }

DEV bfx16 cat16(const unsigned short* lo, const unsigned short* hi) {
  bfx8 a = *(const bfx8*)lo;
  bfx8 b = *(const bfx8*)hi;
  return __builtin_shufflevector(a, b, 0,1,2,3,4,5,6,7,8,9,10,11,12,13,14,15);
}

// ---------------- async global->LDS copy (CDNA5), with safe fallback ----------------
#if __has_builtin(__builtin_amdgcn_global_load_async_to_lds_b128)
#define HAVE_ASYNC_LDS 1
#else
#define HAVE_ASYNC_LDS 0
#endif

DEV void cp16(const unsigned short* g, unsigned short* l) {
#if HAVE_ASYNC_LDS
  __builtin_amdgcn_global_load_async_to_lds_b128(
      (__attribute__((address_space(1))) v4i*)g,
      (__attribute__((address_space(3))) v4i*)l, 0, 0);
#else
  *(uint4*)l = *(const uint4*)g;
#endif
}

DEV void wait_async() {
#if HAVE_ASYNC_LDS
#if __has_builtin(__builtin_amdgcn_s_wait_asynccnt)
  __builtin_amdgcn_s_wait_asynccnt(0);
#else
  asm volatile("s_wait_asynccnt 0" ::: "memory");
#endif
#endif
}

// ---------------- fp32 -> bf16 conversion ----------------
__global__ void k_f32_to_bf16(const float* __restrict__ src,
                              unsigned short* __restrict__ dst, int n) {
  int i = blockIdx.x * blockDim.x + threadIdx.x;
  if (i < n) dst[i] = f2bf(src[i]);
}

// ---------------- Tiled bf16 WMMA GEMM: C[M,N] = A[M,K] * B[N,K]^T + bias ----------------
// Double-buffered LDS; staging via GLOBAL_LOAD_ASYNC_TO_LDS_B128 overlapped with WMMA.
template <bool OUT_F32>
__global__ __launch_bounds__(256)
void k_gemm(const unsigned short* __restrict__ A, const unsigned short* __restrict__ Bm,
            const float* __restrict__ bias, void* __restrict__ Cout,
            int M, int N, int K) {
  constexpr int BM = 128, BN = 128, BK = 32, LSTR = BK + 8; // pad to kill bank conflicts
  __shared__ __align__(16) unsigned short As[2][BM * LSTR];
  __shared__ __align__(16) unsigned short Bs[2][BN * LSTR];

  const int tid  = threadIdx.x;
  const int wid  = tid >> 5;
  const int lane = tid & 31;
  const int lh   = (lane < 16) ? 0 : 1;   // lane half
  const int lr   = lane & 15;

  const int m_blk = blockIdx.x * BM;
  const int n_blk = blockIdx.y * BN;
  const int wm = (wid >> 2) * 64;         // wave row  (2 wave-rows)
  const int wn = (wid & 3) * 32;          // wave col  (4 wave-cols)

  const int ldr = tid >> 1;               // staging row 0..127
  const int ldc = (tid & 1) * 16;         // staging col 0 / 16

  const unsigned short* gA = A + (size_t)(m_blk + ldr) * K + ldc;
  const unsigned short* gB = Bm + (size_t)(n_blk + ldr) * K + ldc;
  unsigned short* lA0 = &As[0][ldr * LSTR + ldc];
  unsigned short* lB0 = &Bs[0][ldr * LSTR + ldc];
  unsigned short* lA1 = &As[1][ldr * LSTR + ldc];
  unsigned short* lB1 = &Bs[1][ldr * LSTR + ldc];

  fx8 acc[4][2] = {};

  // prologue: stage tile 0
  cp16(gA, lA0);     cp16(gA + 8, lA0 + 8);
  cp16(gB, lB0);     cp16(gB + 8, lB0 + 8);
  wait_async();
  __syncthreads();

  const int nk = K / BK;
  for (int t = 0; t < nk; ++t) {
    const int cur = t & 1;
    if (t + 1 < nk) {              // prefetch next tile into the other buffer
      const unsigned short* a = gA + (size_t)(t + 1) * BK;
      const unsigned short* b = gB + (size_t)(t + 1) * BK;
      unsigned short* la = cur ? lA0 : lA1;
      unsigned short* lb = cur ? lB0 : lB1;
      cp16(a, la);   cp16(a + 8, la + 8);
      cp16(b, lb);   cp16(b + 8, lb + 8);
    }

    bfx16 af[4];
#pragma unroll
    for (int ms = 0; ms < 4; ++ms) {
      const unsigned short* p = &As[cur][(wm + ms * 16 + lr) * LSTR];
      af[ms] = cat16(p + (lh ? 8 : 0), p + (lh ? 24 : 16));
    }
    bfx16 bfr[2];
#pragma unroll
    for (int ns = 0; ns < 2; ++ns) {
      const unsigned short* p = &Bs[cur][(wn + ns * 16 + lr) * LSTR];
      bfr[ns] = cat16(p + (lh ? 8 : 0), p + (lh ? 24 : 16));
    }
#pragma unroll
    for (int ms = 0; ms < 4; ++ms)
#pragma unroll
      for (int ns = 0; ns < 2; ++ns)
        acc[ms][ns] = __builtin_amdgcn_wmma_f32_16x16x32_bf16(
            false, af[ms], false, bfr[ns], (short)0, acc[ms][ns], false, false);

    wait_async();
    __syncthreads();
  }

  const int lh8 = lh ? 8 : 0;
#pragma unroll
  for (int ms = 0; ms < 4; ++ms) {
#pragma unroll
    for (int ns = 0; ns < 2; ++ns) {
      const int col = n_blk + wn + ns * 16 + lr;
      const float bv = bias[col];
#pragma unroll
      for (int r = 0; r < 8; ++r) {
        const int row = m_blk + wm + ms * 16 + r + lh8;
        const float vv = acc[ms][ns][r] + bv;
        if (OUT_F32) ((float*)Cout)[(size_t)row * N + col] = vv;
        else ((unsigned short*)Cout)[(size_t)row * N + col] = f2bf(vv);
      }
    }
  }
}

// ---------------- partial RoPE + repack to [B,NH,S,96] ----------------
__global__ void k_rope_pack(const unsigned short* __restrict__ qkv,
                            const float* __restrict__ cosb, const float* __restrict__ sinb,
                            unsigned short* __restrict__ q, unsigned short* __restrict__ k,
                            unsigned short* __restrict__ v) {
  const int t = blockIdx.x;         // 0..T-1
  const int h = blockIdx.y;         // 0..NH-1
  const int d = threadIdx.x;        // 0..95
  const int s = t & (Sc - 1);
  const int b = t >> 10;
  const size_t obase = (((size_t)(b * NHc + h)) * Sc + s) * DP + d;
  const size_t ibase = (size_t)t * QKV_N;

#pragma unroll
  for (int which = 0; which < 2; ++which) {    // 0=q, 1=k
    const size_t off = ibase + (size_t)which * (NHc * HSc) + h * HSc;
    float val;
    if (d < ROTc) {
      const int i = d & 15;
      const float x1 = bf2f(qkv[off + i]);
      const float x2 = bf2f(qkv[off + i + 16]);
      const float c  = cosb[s * 16 + i];
      const float sn = sinb[s * 16 + i];
      val = (d < 16) ? (x1 * c - x2 * sn) : (x2 * c + x1 * sn);
    } else if (d < HSc) {
      val = bf2f(qkv[off + d]);
    } else {
      val = 0.f;
    }
    (which ? k : q)[obase] = f2bf(val);
  }
  {
    const size_t off = ibase + 2 * (size_t)(NHc * HSc) + h * HSc;
    v[obase] = (d < HSc) ? qkv[off + d] : (unsigned short)0;
  }
}

// ---------------- flash attention: 1 WG per (b, h, 64-row q-tile) ----------------
__global__ __launch_bounds__(128)
void k_attn(const unsigned short* __restrict__ q, const unsigned short* __restrict__ k,
            const unsigned short* __restrict__ v, unsigned short* __restrict__ o) {
  constexpr int QT = 64, KT = 32;
  constexpr int KSTR = DP + 8;   // 104
  constexpr int VSTR = KT + 8;   // 40
  constexpr int PSTR = KT + 8;   // 40
  __shared__ __align__(16) unsigned short Ks[KT * KSTR];       // keys   [32][104]
  __shared__ __align__(16) unsigned short Vt[DP * VSTR];       // V^T    [96][40]
  __shared__ __align__(16) unsigned short Ps[4 * 16 * PSTR];   // per-wave P [16][40]

  const int tid = threadIdx.x, wv = tid >> 5, lane = tid & 31;
  const int lh = (lane < 16) ? 0 : 1, lr = lane & 15;
  const int qt = blockIdx.x & 15;
  const int h  = (blockIdx.x >> 4) & 31;
  const int b  = blockIdx.x >> 9;
  const size_t head_base = ((size_t)(b * NHc + h)) * Sc * DP;
  const int q0w = qt * QT + wv * 16;   // this wave's first q row

  // Q fragments (3 K-chunks of 32 over padded head dim 96) — kept in registers
  bfx16 qf[3];
#pragma unroll
  for (int c = 0; c < 3; ++c) {
    const unsigned short* p = q + head_base + (size_t)(q0w + lr) * DP + c * 32;
    qf[c] = cat16(p + (lh ? 8 : 0), p + (lh ? 24 : 16));
  }

  fx8 Oa[6] = {};
  float mrow[8], lrow[8];
#pragma unroll
  for (int r = 0; r < 8; ++r) { mrow[r] = -3.0e38f; lrow[r] = 0.f; }

  const int nkt = 2 * (qt + 1);        // causal: only tiles up to the diagonal
  for (int kt = 0; kt < nkt; ++kt) {
    // stage K tile [32][96] via async LDS copies
#pragma unroll
    for (int i = 0; i < 3; ++i) {
      const int c = tid + i * 128;               // chunk of 8 elems, 384 chunks
      const int row = c / 12, col = (c % 12) * 8;
      cp16(k + head_base + (size_t)(kt * KT + row) * DP + col, &Ks[row * KSTR + col]);
    }
    // stage V transposed: Vt[d][s]
#pragma unroll
    for (int i = 0; i < 3; ++i) {
      const int c = tid + i * 128;
      const int row = c / 12, col = (c % 12) * 8;
      const unsigned short* g = v + head_base + (size_t)(kt * KT + row) * DP + col;
      uint4 x = *(const uint4*)g;
      const unsigned short* xs = (const unsigned short*)&x;
#pragma unroll
      for (int e = 0; e < 8; ++e) Vt[(col + e) * VSTR + row] = xs[e];
    }
    wait_async();
    __syncthreads();

    if (kt * KT <= q0w + 15) {           // tile not fully masked for this wave
      // ---- scores: S(16x32) = Q * K^T ----
      fx8 sa[2] = {};
#pragma unroll
      for (int ns = 0; ns < 2; ++ns)
#pragma unroll
        for (int c = 0; c < 3; ++c) {
          const unsigned short* p = &Ks[(ns * 16 + lr) * KSTR + c * 32];
          bfx16 bfr = cat16(p + (lh ? 8 : 0), p + (lh ? 24 : 16));
          sa[ns] = __builtin_amdgcn_wmma_f32_16x16x32_bf16(
              false, qf[c], false, bfr, (short)0, sa[ns], false, false);
        }

      // ---- scale, causal mask, online softmax ----
#pragma unroll
      for (int r = 0; r < 8; ++r) {
        const int rowg = q0w + r + (lh ? 8 : 0);
        float s0 = sa[0][r] * SCALEc, s1 = sa[1][r] * SCALEc;
        const int c0 = kt * KT + lr;
        if (c0 > rowg)      s0 = -3.0e38f;
        if (c0 + 16 > rowg) s1 = -3.0e38f;
        float mx = fmaxf(s0, s1);
#pragma unroll
        for (int off = 1; off < 16; off <<= 1) mx = fmaxf(mx, __shfl_xor(mx, off, 32));
        const float mo = mrow[r];
        const float mn = fmaxf(mo, mx);
        const float alpha = __expf(mo - mn);
        const float p0 = __expf(s0 - mn);
        const float p1 = __expf(s1 - mn);
        float ps = p0 + p1;
#pragma unroll
        for (int off = 1; off < 16; off <<= 1) ps += __shfl_xor(ps, off, 32);
        mrow[r] = mn;
        lrow[r] = lrow[r] * alpha + ps;
#pragma unroll
        for (int n = 0; n < 6; ++n) Oa[n][r] *= alpha;
        const int prow = r + (lh ? 8 : 0);
        Ps[(wv * 16 + prow) * PSTR + lr]      = f2bf(p0);
        Ps[(wv * 16 + prow) * PSTR + 16 + lr] = f2bf(p1);
      }

      // ---- O += P * V  (P round-tripped through per-wave LDS to A-layout) ----
      const unsigned short* pp = &Ps[(wv * 16 + lr) * PSTR];
      bfx16 pf = cat16(pp + (lh ? 8 : 0), pp + (lh ? 24 : 16));
#pragma unroll
      for (int n = 0; n < 6; ++n) {
        const unsigned short* bp = &Vt[(n * 16 + lr) * VSTR];
        bfx16 bfr = cat16(bp + (lh ? 8 : 0), bp + (lh ? 24 : 16));
        Oa[n] = __builtin_amdgcn_wmma_f32_16x16x32_bf16(
            false, pf, false, bfr, (short)0, Oa[n], false, false);
      }
    }
    __syncthreads();
  }

  // epilogue: O /= l, pack to [T, NH*80]
#pragma unroll
  for (int r = 0; r < 8; ++r) {
    const int rowg = q0w + r + (lh ? 8 : 0);
    const size_t trow = (size_t)(b * Sc + rowg) * Hc + h * HSc;
    const float inv = 1.0f / lrow[r];
#pragma unroll
    for (int n = 0; n < 6; ++n) {
      const int co = n * 16 + lr;
      if (co < HSc) o[trow + co] = f2bf(Oa[n][r] * inv);
    }
  }
}

// ---------------- launch ----------------
extern "C" void kernel_launch(void* const* d_in, const int* in_sizes, int n_in,
                              void* d_out, int out_size, void* d_ws, size_t ws_size,
                              hipStream_t stream) {
  (void)in_sizes; (void)n_in; (void)out_size; (void)ws_size;
  const float* hs      = (const float*)d_in[0];
  const float* cosb    = (const float*)d_in[1];
  const float* sinb    = (const float*)d_in[2];
  const float* w_qkv   = (const float*)d_in[3];
  const float* b_qkv   = (const float*)d_in[4];
  const float* w_dense = (const float*)d_in[5];
  const float* b_dense = (const float*)d_in[6];
  float* out = (float*)d_out;
  char* ws = (char*)d_ws;

  // workspace layout (bytes, all 256-aligned)
  constexpr size_t off_hs     = 0;                               // T*H bf16 (reused as attn_out)
  constexpr size_t off_wqkv   = off_hs     + (size_t)Tc * Hc * 2;
  constexpr size_t off_wdense = off_wqkv   + (size_t)QKV_N * Hc * 2;
  constexpr size_t off_qkv    = off_wdense + (size_t)Hc * Hc * 2;
  constexpr size_t off_q      = off_qkv    + (size_t)Tc * QKV_N * 2;
  constexpr size_t off_k      = off_q      + (size_t)Bc * NHc * Sc * DP * 2;
  constexpr size_t off_v      = off_k      + (size_t)Bc * NHc * Sc * DP * 2;

  unsigned short* hs_bf     = (unsigned short*)(ws + off_hs);
  unsigned short* wqkv_bf   = (unsigned short*)(ws + off_wqkv);
  unsigned short* wdense_bf = (unsigned short*)(ws + off_wdense);
  unsigned short* qkv_bf    = (unsigned short*)(ws + off_qkv);
  unsigned short* qpad      = (unsigned short*)(ws + off_q);
  unsigned short* kpad      = (unsigned short*)(ws + off_k);
  unsigned short* vpad      = (unsigned short*)(ws + off_v);
  unsigned short* attn_bf   = hs_bf;   // hidden bf16 dead after QKV GEMM

  // 1) fp32 -> bf16
  { int n = Tc * Hc;
    k_f32_to_bf16<<<(n + 255) / 256, 256, 0, stream>>>(hs, hs_bf, n); }
  { int n = QKV_N * Hc;
    k_f32_to_bf16<<<(n + 255) / 256, 256, 0, stream>>>(w_qkv, wqkv_bf, n); }
  { int n = Hc * Hc;
    k_f32_to_bf16<<<(n + 255) / 256, 256, 0, stream>>>(w_dense, wdense_bf, n); }

  // 2) QKV GEMM (bf16 out)
  { dim3 g(Tc / 128, QKV_N / 128);
    k_gemm<false><<<g, 256, 0, stream>>>(hs_bf, wqkv_bf, b_qkv, qkv_bf, Tc, QKV_N, Hc); }

  // 3) RoPE + repack
  { dim3 g(Tc, NHc);
    k_rope_pack<<<g, 96, 0, stream>>>(qkv_bf, cosb, sinb, qpad, kpad, vpad); }

  // 4) flash attention
  k_attn<<<Bc * NHc * (Sc / 64), 128, 0, stream>>>(qpad, kpad, vpad, attn_bf);

  // 5) dense projection (fp32 out)
  { dim3 g(Tc / 128, Hc / 128);
    k_gemm<true><<<g, 256, 0, stream>>>(attn_bf, wdense_bf, b_dense, out, Tc, Hc, Hc); }
}
